// LSTM1_19292993093750
// MI455X (gfx1250) — compile-verified, matrix operand order checked by reading, not run on hardware
//
#include <hip/hip_runtime.h>

// Graves handwriting LSTM + GMM attention window for MI455X (gfx1250).
// Persistent kernel: 8 workgroups x 16 batch rows, 600 sequential steps.
// Gates GEMM *and* window-param GEMM run on v_wmma_f32_16x16x32_bf16;
// c-state stays in VGPRs for the whole sequence.

typedef __bf16 bf16;
typedef __attribute__((ext_vector_type(16))) __bf16 v16bf;
typedef __attribute__((ext_vector_type(8)))  __bf16 v8bf;
typedef __attribute__((ext_vector_type(8)))  float  v8f;

#define BB 128
#define TT 600
#define LL 64
#define VV 60
#define HH 400
#define KK 10
#define KDIM 480        // padded gates-GEMM K (3+60+1pad + 400 + 16pad)
#define KSTEPS 15       // 480 / 32
#define NTILES 100      // 1600 / 16
#define JT 25           // 400 / 16 hidden-unit tiles
#define WKSTEPS 13      // window GEMM: 416 / 32 (h K-range 64..479 of act)
#define ACT_STRIDE 488  // padded LDS K-stride (bf16 elems) -> conflict-free ds_load_b128
#define MB 16           // batch rows per workgroup
#define NWAVE 16
#define NTHR 512
#define WPACK_ELEMS  ((size_t)NTILES * KSTEPS * 32 * 16)
#define WPACK2_ELEMS ((size_t)2 * WKSTEPS * 32 * 16)

__device__ __forceinline__ float sigmf(float z) { return 1.f / (1.f + __expf(-z)); }

// Pack [W_ih | 0 | W_hh | 0] (1600 x 480) f32 -> bf16, fragment-linear B-layout:
// fragment (tile, ks, lane) = 16 contiguous bf16 = lane's v16bf for the WMMA B operand.
__global__ void pack_weights(const float* __restrict__ W_ih,
                             const float* __restrict__ W_hh,
                             bf16* __restrict__ Wpack) {
    int tid = blockIdx.x * blockDim.x + threadIdx.x;
    if (tid >= NTILES * KSTEPS * 32) return;
    int tile = tid / (KSTEPS * 32);
    int rem  = tid % (KSTEPS * 32);
    int ks   = rem / 32;
    int lane = rem % 32;
    int n    = tile * 16 + (lane & 15);          // gate column 0..1599
    int kb   = ks * 32 + ((lane >> 4) & 1) * 8;  // lane-half K offset
    bf16* dst = Wpack + (size_t)tid * 16;
#pragma unroll
    for (int i = 0; i < 16; ++i) {
        int kk = kb + (i < 8 ? i : i + 8);       // K = kb+{0..7}, kb+{16..23}
        float v = 0.f;
        if (kk < 63)                   v = W_ih[n * 63 + kk];
        else if (kk >= 64 && kk < 464) v = W_hh[n * HH + (kk - 64)];
        dst[i] = (bf16)v;
    }
}

// Pack W_win (30 x 400) -> bf16 fragment-linear layout for the window GEMM.
// B-matrix K index kk maps to act K offset 64+kk (where h lives), j = kk.
__global__ void pack_win(const float* __restrict__ W_win, bf16* __restrict__ Wpack2) {
    int tid = blockIdx.x * blockDim.x + threadIdx.x;
    if (tid >= 2 * WKSTEPS * 32) return;
    int tile = tid / (WKSTEPS * 32);
    int rem  = tid % (WKSTEPS * 32);
    int ks   = rem / 32;
    int lane = rem % 32;
    int p    = tile * 16 + (lane & 15);          // param column 0..31 (30 valid)
    int kb   = ks * 32 + ((lane >> 4) & 1) * 8;
    bf16* dst = Wpack2 + (size_t)tid * 16;
#pragma unroll
    for (int i = 0; i < 16; ++i) {
        int j = kb + (i < 8 ? i : i + 8);        // h index 0..415 (400 valid)
        float v = (p < 30 && j < HH) ? W_win[p * HH + j] : 0.f;
        dst[i] = (bf16)v;
    }
}

__global__ void pack_bias(const float* __restrict__ b_ih,
                          const float* __restrict__ b_hh,
                          float* __restrict__ bsum) {
    int tid = blockIdx.x * blockDim.x + threadIdx.x;
    if (tid < 4 * HH) bsum[tid] = b_ih[tid] + b_hh[tid];
}

__global__ __launch_bounds__(NTHR)
void lstm_attn_kernel(const float* __restrict__ x,
                      const float* __restrict__ onehots,
                      const float* __restrict__ text_lens,
                      const float* __restrict__ w0,
                      const float* __restrict__ kappa0,
                      const float* __restrict__ h0,
                      const float* __restrict__ c0,
                      const float* __restrict__ b_win,
                      const bf16*  __restrict__ Wpack,
                      const bf16*  __restrict__ Wpack2,
                      const float* __restrict__ bsum,
                      float* __restrict__ ws_out,
                      float* __restrict__ hs_out,
                      float* __restrict__ h_out,
                      float* __restrict__ c_out,
                      float* __restrict__ w_out,
                      float* __restrict__ kap_out,
                      float* __restrict__ phi_out) {
    __shared__ bf16  act[MB * ACT_STRIDE];   // bf16 activations [x|w|pad|h|pad]
    __shared__ float par[MB * 32];           // exp(window params), 30 used
    __shared__ float kls[MB * KK];           // kappa state
    __shared__ float phl[MB * 66];           // phi, 65 used
    __shared__ float scl[MB];                // L / text_len

    const int tid  = threadIdx.x;
    const int lane = tid & 31;
    const int wave = tid >> 5;
    const int rb   = blockIdx.x * MB;

    // ---------------- init ----------------
    for (int i = tid; i < MB * ACT_STRIDE; i += NTHR) act[i] = (bf16)0.f;
    __syncthreads();
    for (int i = tid; i < MB * HH; i += NTHR) {
        int row = i / HH, j = i % HH;
        act[row * ACT_STRIDE + 64 + j] = (bf16)h0[(rb + row) * HH + j];
    }
    for (int i = tid; i < MB * VV; i += NTHR) {
        int row = i / VV, v = i % VV;
        act[row * ACT_STRIDE + 3 + v] = (bf16)w0[(rb + row) * VV + v];
    }
    for (int i = tid; i < MB * KK; i += NTHR) kls[i] = kappa0[rb * KK + i];
    if (tid < MB) scl[tid] = (float)LL / text_lens[rb + tid];

    // c-state lives in registers for the whole sequence (C/D fragment layout)
    float creg[2][8];
#pragma unroll
    for (int gi = 0; gi < 2; ++gi) {
        int g = wave + NWAVE * gi;
        if (g < JT) {
            int col   = g * 16 + (lane & 15);
            int rbase = (lane >> 4) << 3;
#pragma unroll
            for (int r = 0; r < 8; ++r)
                creg[gi][r] = c0[(rb + rbase + r) * HH + col];
        }
    }
    __syncthreads();

    const int arow  = lane & 15;
    const int ahalf = ((lane >> 4) & 1) * 8;

    // ---------------- time loop ----------------
    for (int t = 0; t < TT; ++t) {
        // phase 1: stage x_t into activation tile
        if (tid < MB * 3) {
            int row = tid / 3, k = tid % 3;
            act[row * ACT_STRIDE + k] = (bf16)x[((rb + row) * TT + t) * 3 + k];
        }
        __syncthreads();

        // phase 2a: gates[16 x 1600] = act[16 x 480] @ W^T via bf16 WMMA
        v8f acc[2][4];
#pragma unroll
        for (int gi = 0; gi < 2; ++gi) {
            int g = wave + NWAVE * gi;
            if (g >= JT) continue;
#pragma unroll
            for (int q = 0; q < 4; ++q) acc[gi][q] = (v8f){0.f,0.f,0.f,0.f,0.f,0.f,0.f,0.f};
            for (int ks = 0; ks < KSTEPS; ++ks) {
                const bf16* ap = &act[arow * ACT_STRIDE + ks * 32 + ahalf];
                v8bf alo = *(const v8bf*)ap;
                v8bf ahi = *(const v8bf*)(ap + 16);
                v16bf afrag = __builtin_shufflevector(alo, ahi,
                    0,1,2,3,4,5,6,7,8,9,10,11,12,13,14,15);
#pragma unroll
                for (int q = 0; q < 4; ++q) {           // i, f, g, o gate tiles
                    int tile = g + q * JT;
                    const bf16* bp = Wpack + (((size_t)(tile * KSTEPS + ks)) * 32 + lane) * 16;
                    v16bf bfrag = *(const v16bf*)bp;    // two global_load_b128
                    acc[gi][q] = __builtin_amdgcn_wmma_f32_16x16x32_bf16(
                        false, afrag, false, bfrag, (short)0, acc[gi][q], false, false);
                }
            }
        }
        __syncthreads();   // all WMMA reads of act done before h is rewritten

        // phase 2b: cell update in registers; emit h (bf16 into act, f32 to global)
#pragma unroll
        for (int gi = 0; gi < 2; ++gi) {
            int g = wave + NWAVE * gi;
            if (g >= JT) continue;
            int col   = g * 16 + (lane & 15);
            int rbase = (lane >> 4) << 3;
            float bi = bsum[col];
            float bf = bsum[HH + col];
            float bg = bsum[2 * HH + col];
            float bo = bsum[3 * HH + col];
#pragma unroll
            for (int r = 0; r < 8; ++r) {
                float iv = sigmf(acc[gi][0][r] + bi);
                float fv = sigmf(acc[gi][1][r] + bf);
                float gv = tanhf(acc[gi][2][r] + bg);
                float ov = sigmf(acc[gi][3][r] + bo);
                float cv = fv * creg[gi][r] + iv * gv;
                creg[gi][r] = cv;
                float hv = ov * tanhf(cv);
                int row = rbase + r;
                act[row * ACT_STRIDE + 64 + col] = (bf16)hv;
                hs_out[((size_t)(rb + row) * TT + t) * HH + col] = hv;
                if (t == TT - 1) {
                    h_out[(rb + row) * HH + col] = hv;
                    c_out[(rb + row) * HH + col] = cv;
                }
            }
        }
        __syncthreads();

        // phase 3: window params = exp(h @ W_win^T + b_win) as WMMA:
        // [16 x 416] @ [416 x 32] -> waves 0,1 each own one 16-col param tile.
        if (wave < 2) {
            v8f pacc = (v8f){0.f,0.f,0.f,0.f,0.f,0.f,0.f,0.f};
            for (int ks = 0; ks < WKSTEPS; ++ks) {
                const bf16* ap = &act[arow * ACT_STRIDE + 64 + ks * 32 + ahalf];
                v8bf alo = *(const v8bf*)ap;
                v8bf ahi = *(const v8bf*)(ap + 16);
                v16bf afrag = __builtin_shufflevector(alo, ahi,
                    0,1,2,3,4,5,6,7,8,9,10,11,12,13,14,15);
                const bf16* bp = Wpack2 + ((size_t)(wave * WKSTEPS + ks) * 32 + lane) * 16;
                v16bf bfrag = *(const v16bf*)bp;
                pacc = __builtin_amdgcn_wmma_f32_16x16x32_bf16(
                    false, afrag, false, bfrag, (short)0, pacc, false, false);
            }
            int p     = wave * 16 + (lane & 15);
            int rbase = (lane >> 4) << 3;
            if (p < 30) {
                float bw = b_win[p];
#pragma unroll
                for (int r = 0; r < 8; ++r)
                    par[(rbase + r) * 32 + p] = __expf(pacc[r] + bw);
            }
        }
        __syncthreads();

        // phase 4: kappa += exp(pre_kappa); then phi[row][l]
        if (tid < MB * KK) {
            int row = tid / KK, k = tid % KK;
            float kap = kls[tid] + par[row * 32 + 20 + k];
            kls[tid] = kap;
            if (t == TT - 1) kap_out[rb * KK + tid] = kap;
        }
        __syncthreads();
        for (int idx = tid; idx < MB * 65; idx += NTHR) {
            int row = idx / 65, l = idx % 65;
            float s = 0.f;
#pragma unroll
            for (int k = 0; k < KK; ++k) {
                float d = kls[row * KK + k] - (float)l;
                s += par[row * 32 + k] * __expf(-par[row * 32 + 10 + k] * d * d);
            }
            float ph = s * scl[row];
            phl[row * 66 + l] = ph;
            if (t == TT - 1) phi_out[(rb + row) * 65 + l] = ph;
        }
        __syncthreads();

        // phase 5: w[row][v] = sum_l phi[l] * onehot[l][v]; float4-vectorized
        for (int idx = tid; idx < MB * (VV / 4); idx += NTHR) {
            int row = idx / (VV / 4), v4 = (idx % (VV / 4)) * 4;
            float4 s = make_float4(0.f, 0.f, 0.f, 0.f);
            const float* oh = &onehots[(size_t)(rb + row) * LL * VV + v4];
            const float* pp = &phl[row * 66];
#pragma unroll 4
            for (int l = 0; l < LL; ++l) {
                float  ph = pp[l];
                float4 o  = *(const float4*)(oh + l * VV);
                s.x += ph * o.x; s.y += ph * o.y; s.z += ph * o.z; s.w += ph * o.w;
            }
            *(float4*)&ws_out[((size_t)(rb + row) * TT + t) * VV + v4] = s;
            bf16* aw = &act[row * ACT_STRIDE + 3 + v4];
            aw[0] = (bf16)s.x; aw[1] = (bf16)s.y; aw[2] = (bf16)s.z; aw[3] = (bf16)s.w;
            if (t == TT - 1) *(float4*)&w_out[(rb + row) * VV + v4] = s;
        }
        __syncthreads();
    }
}

extern "C" void kernel_launch(void* const* d_in, const int* in_sizes, int n_in,
                              void* d_out, int out_size, void* d_ws, size_t ws_size,
                              hipStream_t stream) {
    const float* x         = (const float*)d_in[0];
    const float* onehots   = (const float*)d_in[1];
    const float* text_lens = (const float*)d_in[2];
    const float* w0        = (const float*)d_in[3];
    const float* kappa0    = (const float*)d_in[4];
    const float* h0        = (const float*)d_in[5];
    const float* c0        = (const float*)d_in[6];
    const float* W_ih      = (const float*)d_in[7];
    const float* b_ih      = (const float*)d_in[8];
    const float* W_hh      = (const float*)d_in[9];
    const float* b_hh      = (const float*)d_in[10];
    const float* W_win     = (const float*)d_in[11];
    const float* b_win     = (const float*)d_in[12];

    bf16*  Wpack  = (bf16*)d_ws;
    float* bsum   = (float*)((char*)d_ws + WPACK_ELEMS * sizeof(bf16));
    bf16*  Wpack2 = (bf16*)((char*)d_ws + WPACK_ELEMS * sizeof(bf16) + 4 * HH * sizeof(float));

    int packThreads = NTILES * KSTEPS * 32;
    pack_weights<<<(packThreads + 255) / 256, 256, 0, stream>>>(W_ih, W_hh, Wpack);
    pack_win<<<(2 * WKSTEPS * 32 + 255) / 256, 256, 0, stream>>>(W_win, Wpack2);
    pack_bias<<<(4 * HH + 255) / 256, 256, 0, stream>>>(b_ih, b_hh, bsum);

    float* out     = (float*)d_out;
    float* ws_o    = out;                                   // [B,T,V]
    float* hs_o    = ws_o + (size_t)BB * TT * VV;           // [B,T,H]
    float* h_o     = hs_o + (size_t)BB * TT * HH;           // [B,H]
    float* c_o     = h_o + BB * HH;                         // [B,H]
    float* w_o     = c_o + BB * HH;                         // [B,V]
    float* kap_o   = w_o + BB * VV;                         // [B,K]
    float* phi_o   = kap_o + BB * KK;                       // [B,L+1]

    lstm_attn_kernel<<<BB / MB, NTHR, 0, stream>>>(
        x, onehots, text_lens, w0, kappa0, h0, c0, b_win,
        Wpack, Wpack2, bsum, ws_o, hs_o, h_o, c_o, w_o, kap_o, phi_o);
}
